// SpectralConv2d_46480136077788
// MI455X (gfx1250) — compile-verified, matrix operand order checked by reading, not run on hardware
//
#include <hip/hip_runtime.h>
#include <hip/hip_bf16.h>

// ---------------------------------------------------------------------------
// Problem constants
// ---------------------------------------------------------------------------
#define BB     8
#define CC     64
#define NN     2048
#define MM     32                    // M_MODES
#define MFREQ  4032                  // 2m*(2m-1)
#define BC     (BB * CC)             // 512
#define NMODE  (MM * MM)             // 1024 (x,y) mode pairs
#define EMB    256

// ---------------------------------------------------------------------------
// WMMA types (CDNA5 gfx1250, wave32)
// ---------------------------------------------------------------------------
typedef __bf16 bf16x16 __attribute__((ext_vector_type(16)));
typedef __bf16 bf16x8  __attribute__((ext_vector_type(8)));
typedef float  f32x8   __attribute__((ext_vector_type(8)));

// float -> bf16 bits, round-to-nearest-even
__device__ __forceinline__ unsigned short f2bf(float f) {
    union { float f; unsigned u; } v; v.f = f;
    unsigned r = v.u + 0x7FFFu + ((v.u >> 16) & 1u);
    return (unsigned short)(r >> 16);
}

// Load one 16x32 bf16 WMMA operand fragment (A-style layout; the B operand of
// D = A * X^T uses the identical per-lane pattern) straight from a row-major
// K-contiguous matrix.  Per ISA 7.12.2: lane<16 holds K {0..7, 16..23},
// lane>=16 holds K {8..15, 24..31}  -> two b128 loads per lane.
__device__ __forceinline__ bf16x16 load_frag(const unsigned short* __restrict__ base,
                                             int row, long ld, int kBase, int lane) {
    const int l  = lane & 15;
    const int hi = lane >> 4;
    const unsigned short* p = base + (long)(row + l) * ld + kBase + hi * 8;
    union { bf16x16 v; bf16x8 h[2]; } u;
    u.h[0] = *(const bf16x8*)(p);
    u.h[1] = *(const bf16x8*)(p + 16);
    return u.v;
}

// ---------------------------------------------------------------------------
// fp32 -> bf16 (with scale; scale=-1 folds the Hermitian minus into Vi_im)
// ---------------------------------------------------------------------------
__global__ __launch_bounds__(256)
void f32_to_bf16(const float* __restrict__ in, unsigned short* __restrict__ out,
                 long n, float scale) {
    long i      = ((long)blockIdx.x * blockDim.x + threadIdx.x) * 4;
    long stride = (long)gridDim.x * blockDim.x * 4;
    for (; i < n; i += stride) {
        float4 v = *(const float4*)(in + i);
        unsigned long long pk =
            (unsigned long long)f2bf(v.x * scale)
          | ((unsigned long long)f2bf(v.y * scale) << 16)
          | ((unsigned long long)f2bf(v.z * scale) << 32)
          | ((unsigned long long)f2bf(v.w * scale) << 48);
        *(unsigned long long*)(out + i) = pk;
    }
}

// ---------------------------------------------------------------------------
// Transpose w[i][o][x][y] ([4096 io][1024 xy]) -> wT[mode][i][o] ([1024][4096])
// LDS-tiled so both sides are coalesced.  grid.z selects one of 4 arrays.
// ---------------------------------------------------------------------------
__global__ __launch_bounds__(256)
void transpose_w(const float* __restrict__ w1re, const float* __restrict__ w1im,
                 const float* __restrict__ w2re, const float* __restrict__ w2im,
                 float* __restrict__ t1re, float* __restrict__ t1im,
                 float* __restrict__ t2re, float* __restrict__ t2im) {
    __shared__ float tile[32][33];
    const float* src; float* dst;
    switch (blockIdx.z) {
        case 0:  src = w1re; dst = t1re; break;
        case 1:  src = w1im; dst = t1im; break;
        case 2:  src = w2re; dst = t2re; break;
        default: src = w2im; dst = t2im; break;
    }
    const int xy0 = blockIdx.x * 32;      // 1024 / 32
    const int io0 = blockIdx.y * 32;      // 4096 / 32
    const int tx = threadIdx.x, ty = threadIdx.y;   // (32, 8)
    #pragma unroll
    for (int j = 0; j < 32; j += 8)
        tile[ty + j][tx] = src[(long)(io0 + ty + j) * 1024 + xy0 + tx];
    __syncthreads();
    #pragma unroll
    for (int j = 0; j < 32; j += 8)
        dst[(long)(xy0 + ty + j) * 4096 + io0 + tx] = tile[tx][ty + j];
}

// ---------------------------------------------------------------------------
// phi[b][mode] = sum_e m[mode][e] * emb[b][e]   (4 arrays; one wave / output)
// ---------------------------------------------------------------------------
__global__ __launch_bounds__(256)
void phi_kernel(const float* __restrict__ emb,
                const float* __restrict__ m1re, const float* __restrict__ m1im,
                const float* __restrict__ m2re, const float* __restrict__ m2im,
                float* __restrict__ p1r, float* __restrict__ p1i,
                float* __restrict__ p2r, float* __restrict__ p2i) {
    const int wid  = blockIdx.x * 8 + (threadIdx.x >> 5);   // 8192 waves
    const int lane = threadIdx.x & 31;
    const int b    = wid >> 10;
    const int mode = wid & 1023;
    float s1r = 0.f, s1i = 0.f, s2r = 0.f, s2i = 0.f;
    #pragma unroll
    for (int e = lane; e < EMB; e += 32) {
        const float ev = emb[b * EMB + e];
        const long  mi = (long)mode * EMB + e;
        s1r += m1re[mi] * ev;  s1i += m1im[mi] * ev;
        s2r += m2re[mi] * ev;  s2i += m2im[mi] * ev;
    }
    #pragma unroll
    for (int off = 16; off; off >>= 1) {
        s1r += __shfl_down(s1r, off, 32);
        s1i += __shfl_down(s1i, off, 32);
        s2r += __shfl_down(s2r, off, 32);
        s2i += __shfl_down(s2i, off, 32);
    }
    if (lane == 0) {
        const int idx = b * NMODE + mode;
        p1r[idx] = s1r; p1i[idx] = s1i; p2r[idx] = s2r; p2i[idx] = s2i;
    }
}

// ---------------------------------------------------------------------------
// Forward NUFT GEMM:  Y[f][bc] = sum_n Vf[f][n] * X[bc][n]
// bf16 operands, fp32 accum.  One wave computes a 64x32 block (4x2 WMMA tiles:
// 6 fragment loads -> 8 WMMAs per k-step).
// grid = (BC/32, MFREQ/64, 2)   z: 0 -> (Vf_re, Y_re), 1 -> (Vf_im, Y_im)
// ---------------------------------------------------------------------------
__global__ __launch_bounds__(32)
void nuft_fwd_gemm(const unsigned short* __restrict__ VfRe,
                   const unsigned short* __restrict__ VfIm,
                   const unsigned short* __restrict__ X,
                   float* __restrict__ Yre, float* __restrict__ Yim) {
    const int lane = threadIdx.x;
    const unsigned short* A = blockIdx.z ? VfIm : VfRe;
    float* Y                = blockIdx.z ? Yim  : Yre;
    const int m0 = blockIdx.y * 64;   // f
    const int n0 = blockIdx.x * 32;   // bc
    f32x8 acc[4][2] = {};
    #pragma unroll 2
    for (int k = 0; k < NN; k += 32) {
        bf16x16 a[4], b[2];
        #pragma unroll
        for (int i = 0; i < 4; ++i) a[i] = load_frag(A, m0 + 16 * i, NN, k, lane);
        #pragma unroll
        for (int j = 0; j < 2; ++j) b[j] = load_frag(X, n0 + 16 * j, NN, k, lane);
        #pragma unroll
        for (int i = 0; i < 4; ++i)
            #pragma unroll
            for (int j = 0; j < 2; ++j)
                acc[i][j] = __builtin_amdgcn_wmma_f32_16x16x32_bf16(
                    false, a[i], false, b[j], (short)0, acc[i][j], false, false);
    }
    const int l = lane & 15, hi = lane >> 4;
    #pragma unroll
    for (int ti = 0; ti < 4; ++ti)
        #pragma unroll
        for (int tj = 0; tj < 2; ++tj)
            #pragma unroll
            for (int r = 0; r < 8; ++r)
                Y[(long)(m0 + ti * 16 + hi * 8 + r) * BC + n0 + tj * 16 + l] = acc[ti][tj][r];
}

// ---------------------------------------------------------------------------
// Mode mixing + phi modulation (fp32 VALU; 1.5% of FLOPs).
// One 512-thread block per mode (x,y).  Writes bf16 P/Q at [bc][g], g < 2048.
// ---------------------------------------------------------------------------
__global__ __launch_bounds__(512)
void mode_mix(const float* __restrict__ Yre, const float* __restrict__ Yim,
              const float* __restrict__ w1Tre, const float* __restrict__ w1Tim,
              const float* __restrict__ w2Tre, const float* __restrict__ w2Tim,
              const float* __restrict__ p1r, const float* __restrict__ p1i,
              const float* __restrict__ p2r, const float* __restrict__ p2i,
              unsigned short* __restrict__ P, unsigned short* __restrict__ Q) {
    __shared__ float in1r[BC], in1i[BC], in2r[BC], in2i[BC];
    const int mode = blockIdx.x;
    const int x = mode >> 5, y = mode & 31;
    const int f1 = x * 63 + y;                 // rows [0, m)   of x_ft
    const int f2 = (x + MM) * 63 + y;          // rows [m, 2m)  of x_ft
    const int t = threadIdx.x;
    in1r[t] = Yre[(long)f1 * BC + t];  in1i[t] = Yim[(long)f1 * BC + t];
    in2r[t] = Yre[(long)f2 * BC + t];  in2i[t] = Yim[(long)f2 * BC + t];
    __syncthreads();
    const int b = t >> 6, o = t & 63;
    const float* w1r = w1Tre + (long)mode * 4096;
    const float* w1i = w1Tim + (long)mode * 4096;
    const float* w2r = w2Tre + (long)mode * 4096;
    const float* w2i = w2Tim + (long)mode * 4096;
    float o1r = 0.f, o1i = 0.f, o2r = 0.f, o2i = 0.f;
    #pragma unroll 4
    for (int i = 0; i < CC; ++i) {
        const float a1r = in1r[b * CC + i], a1i = in1i[b * CC + i];
        const float a2r = in2r[b * CC + i], a2i = in2i[b * CC + i];
        const float x1r = w1r[i * CC + o],  x1i = w1i[i * CC + o];
        const float x2r = w2r[i * CC + o],  x2i = w2i[i * CC + o];
        o1r += a1r * x1r - a1i * x1i;  o1i += a1r * x1i + a1i * x1r;
        o2r += a2r * x2r - a2i * x2i;  o2i += a2r * x2i + a2i * x2r;
    }
    const int pidx = b * NMODE + mode;
    const float h1r = p1r[pidx], h1i = p1i[pidx];
    const float h2r = p2r[pidx], h2i = p2i[pidx];
    const float z1r = h1r * o1r - h1i * o1i, z1i = h1r * o1i + h1i * o1r;
    const float z2r = h2r * o2r - h2i * o2i, z2i = h2r * o2i + h2i * o2r;
    const long row = (long)(b * CC + o) * MFREQ;
    const int g1 = x * MM + y;              // out_ft reshape [2m][m]
    const int g2 = (x + MM) * MM + y;
    P[row + g1] = f2bf(z1r);  Q[row + g1] = f2bf(z1i);
    P[row + g2] = f2bf(z2r);  Q[row + g2] = f2bf(z2i);
}

// ---------------------------------------------------------------------------
// Hermitian extension: P/Q[bc][2048 + j] = conj(P/Q[(b,63-c)][2047 - j])
// ---------------------------------------------------------------------------
__global__ __launch_bounds__(256)
void hermitian_ext(unsigned short* __restrict__ P, unsigned short* __restrict__ Q) {
    const int bc = blockIdx.x;
    const int b = bc >> 6, c = bc & 63;
    const long dstRow = (long)bc * MFREQ;
    const long srcRow = (long)(b * CC + (63 - c)) * MFREQ;
    for (int j = threadIdx.x; j < MFREQ - 2048; j += 256) {
        const int k = 2048 + j;
        const int g = 2047 - j;                     // in [64, 2047]
        P[dstRow + k] = P[srcRow + g];
        Q[dstRow + k] = Q[srcRow + g] ^ 0x8000u;    // conjugate
    }
}

// ---------------------------------------------------------------------------
// Inverse NUFT GEMM (real part only):
//   out[bc][n] = (2/N) * sum_f ( P[bc][f]*ViRe[n][f] + Q[bc][f]*ViImNeg[n][f] )
// One wave computes a 32x64 block (2x4 WMMA tiles over the complex pair:
// 12 fragment loads -> 16 WMMAs per k-step).
// grid = (NN/64, BC/32)
// ---------------------------------------------------------------------------
__global__ __launch_bounds__(32)
void nuft_inv_gemm(const unsigned short* __restrict__ P,
                   const unsigned short* __restrict__ Q,
                   const unsigned short* __restrict__ ViRe,
                   const unsigned short* __restrict__ ViImNeg,
                   float* __restrict__ out) {
    const int lane = threadIdx.x;
    const int m0 = blockIdx.y * 32;   // bc
    const int n0 = blockIdx.x * 64;   // n
    f32x8 acc[2][4] = {};
    #pragma unroll 1
    for (int k = 0; k < MFREQ; k += 32) {
        bf16x16 p[2], q[2], br[4], bi[4];
        #pragma unroll
        for (int i = 0; i < 2; ++i) {
            p[i] = load_frag(P, m0 + 16 * i, MFREQ, k, lane);
            q[i] = load_frag(Q, m0 + 16 * i, MFREQ, k, lane);
        }
        #pragma unroll
        for (int j = 0; j < 4; ++j) {
            br[j] = load_frag(ViRe,    n0 + 16 * j, MFREQ, k, lane);
            bi[j] = load_frag(ViImNeg, n0 + 16 * j, MFREQ, k, lane);
        }
        #pragma unroll
        for (int i = 0; i < 2; ++i)
            #pragma unroll
            for (int j = 0; j < 4; ++j) {
                acc[i][j] = __builtin_amdgcn_wmma_f32_16x16x32_bf16(
                    false, p[i], false, br[j], (short)0, acc[i][j], false, false);
                acc[i][j] = __builtin_amdgcn_wmma_f32_16x16x32_bf16(
                    false, q[i], false, bi[j], (short)0, acc[i][j], false, false);
            }
    }
    const float s = 2.0f / (float)NN;
    const int l = lane & 15, hi = lane >> 4;
    #pragma unroll
    for (int ti = 0; ti < 2; ++ti)
        #pragma unroll
        for (int tj = 0; tj < 4; ++tj)
            #pragma unroll
            for (int r = 0; r < 8; ++r)
                out[(long)(m0 + ti * 16 + hi * 8 + r) * NN + n0 + tj * 16 + l] = acc[ti][tj][r] * s;
}

// ---------------------------------------------------------------------------
// Host launcher
// ---------------------------------------------------------------------------
extern "C" void kernel_launch(void* const* d_in, const int* in_sizes, int n_in,
                              void* d_out, int out_size, void* d_ws, size_t ws_size,
                              hipStream_t stream) {
    const float* x     = (const float*)d_in[0];
    const float* emb   = (const float*)d_in[1];
    const float* Vf_re = (const float*)d_in[2];
    const float* Vf_im = (const float*)d_in[3];
    const float* Vi_re = (const float*)d_in[4];
    const float* Vi_im = (const float*)d_in[5];
    const float* w1_re = (const float*)d_in[6];
    const float* w1_im = (const float*)d_in[7];
    const float* w2_re = (const float*)d_in[8];
    const float* w2_im = (const float*)d_in[9];
    const float* m1_re = (const float*)d_in[10];
    const float* m1_im = (const float*)d_in[11];
    const float* m2_re = (const float*)d_in[12];
    const float* m2_im = (const float*)d_in[13];
    float* outp = (float*)d_out;

    // Workspace carve-up (all chunks 256B aligned; sizes below are multiples)
    char* ws = (char*)d_ws;
    size_t off = 0;
    auto carve = [&](size_t bytes) { void* p = ws + off; off += (bytes + 255) & ~(size_t)255; return p; };
    unsigned short* usX    = (unsigned short*)carve((size_t)BC * NN * 2);         // 2 MB
    unsigned short* usVfRe = (unsigned short*)carve((size_t)MFREQ * NN * 2);      // 16.5 MB
    unsigned short* usVfIm = (unsigned short*)carve((size_t)MFREQ * NN * 2);
    unsigned short* usViRe = (unsigned short*)carve((size_t)NN * MFREQ * 2);
    unsigned short* usViIm = (unsigned short*)carve((size_t)NN * MFREQ * 2);
    float* Yre   = (float*)carve((size_t)MFREQ * BC * 4);                          // 8.25 MB
    float* Yim   = (float*)carve((size_t)MFREQ * BC * 4);
    float* w1Tre = (float*)carve((size_t)NMODE * 4096 * 4);                        // 16.8 MB
    float* w1Tim = (float*)carve((size_t)NMODE * 4096 * 4);
    float* w2Tre = (float*)carve((size_t)NMODE * 4096 * 4);
    float* w2Tim = (float*)carve((size_t)NMODE * 4096 * 4);
    float* p1r   = (float*)carve((size_t)BB * NMODE * 4);
    float* p1i   = (float*)carve((size_t)BB * NMODE * 4);
    float* p2r   = (float*)carve((size_t)BB * NMODE * 4);
    float* p2i   = (float*)carve((size_t)BB * NMODE * 4);
    unsigned short* Pm = (unsigned short*)carve((size_t)BC * MFREQ * 2);           // 4.1 MB
    unsigned short* Qm = (unsigned short*)carve((size_t)BC * MFREQ * 2);
    (void)ws_size; (void)n_in; (void)in_sizes; (void)out_size;

    // 1. bf16 conversions (fold -1 into Vi_im for the real-part subtraction)
    f32_to_bf16<<<2048, 256, 0, stream>>>(x,     usX,    (long)BC * NN,    1.0f);
    f32_to_bf16<<<4096, 256, 0, stream>>>(Vf_re, usVfRe, (long)MFREQ * NN, 1.0f);
    f32_to_bf16<<<4096, 256, 0, stream>>>(Vf_im, usVfIm, (long)MFREQ * NN, 1.0f);
    f32_to_bf16<<<4096, 256, 0, stream>>>(Vi_re, usViRe, (long)NN * MFREQ, 1.0f);
    f32_to_bf16<<<4096, 256, 0, stream>>>(Vi_im, usViIm, (long)NN * MFREQ, -1.0f);

    // 2. weight transpose [i][o][x][y] -> [mode][i][o]  (coalesced both sides)
    transpose_w<<<dim3(32, 128, 4), dim3(32, 8), 0, stream>>>(
        w1_re, w1_im, w2_re, w2_im, w1Tre, w1Tim, w2Tre, w2Tim);

    // 3. phi modulation factors
    phi_kernel<<<1024, 256, 0, stream>>>(emb, m1_re, m1_im, m2_re, m2_im,
                                         p1r, p1i, p2r, p2i);

    // 4. forward NUFT (two bf16 WMMA GEMMs, 4032x512x2048)
    nuft_fwd_gemm<<<dim3(BC / 32, MFREQ / 64, 2), 32, 0, stream>>>(
        usVfRe, usVfIm, usX, Yre, Yim);

    // 5. per-mode channel mixing + modulation -> bf16 spectrum P/Q
    mode_mix<<<NMODE, 512, 0, stream>>>(Yre, Yim, w1Tre, w1Tim, w2Tre, w2Tim,
                                        p1r, p1i, p2r, p2i, Pm, Qm);

    // 6. Hermitian-symmetric extension to full MFREQ
    hermitian_ext<<<BC, 256, 0, stream>>>(Pm, Qm);

    // 7. inverse NUFT (real part; two fused bf16 WMMA GEMMs, 512x2048x4032)
    nuft_inv_gemm<<<dim3(NN / 64, BC / 32), 32, 0, stream>>>(
        Pm, Qm, usViRe, usViIm, outp);
}
